// SchNetLayer_12335146074582
// MI455X (gfx1250) — matrix operand order repaired and checked by reading, not compiled, not run on hardware
//
#include <hip/hip_runtime.h>
#include <hip/hip_bf16.h>

#define N_ELEC 8192
#define N_NUC  512
#define EMB    256
#define KER    256
#define DF     64
#define MID    128
#define E_EDGES 131072

typedef __attribute__((ext_vector_type(16))) __bf16 v16bf;
typedef __attribute__((ext_vector_type(8)))  float  v8f;

// ---------------- WMMA fragment helpers (wave32, 16x16x32 bf16) -------------
// A (16x32, MxK): lane L (M = L&15): elems 0..7  -> K = kBase + 8*(L>>4) + i
//                                   elems 8..15 -> K = kBase + 16 + 8*(L>>4) + i
static __device__ inline v16bf a_frag_f32(const float* __restrict__ row, int kBase, int half) {
  v16bf a;
  const float* p0 = row + kBase + 8 * half;
  const float* p1 = p0 + 16;
#pragma unroll
  for (int i = 0; i < 8; ++i) { a[i] = (__bf16)p0[i]; a[i + 8] = (__bf16)p1[i]; }
  return a;
}
static __device__ inline v16bf a_frag_bf16(const __bf16* row, int kBase, int half) {
  v16bf a;
  ((uint4*)&a)[0] = *(const uint4*)(row + kBase + 8 * half);
  ((uint4*)&a)[1] = *(const uint4*)(row + kBase + 16 + 8 * half);
  return a;
}
// B (32x16, KxN) from TRANSPOSED weights Wt[N][K]: lane L (N = L&15):
// elems 0..7 -> K = kBase + 8*(L>>4) + i ; elems 8..15 -> K = kBase + 16 + 8*(L>>4) + i
static __device__ inline v16bf b_frag_bf16t(const __bf16* rowN, int kBase, int half) {
  v16bf b;
  const __bf16* p = rowN + kBase + 8 * half;
  ((uint4*)&b)[0] = ((const uint4*)p)[0];
  ((uint4*)&b)[1] = ((const uint4*)(p + 16))[0];
  return b;
}
static __device__ inline v8f wmma_bf16(v16bf a, v16bf b, v8f c) {
  return __builtin_amdgcn_wmma_f32_16x16x32_bf16(false, a, false, b, (short)0, c, false, false);
}
// shifted softplus: log(0.5*e^x + 0.5), numerically stable
static __device__ inline float ssp(float x) {
  return fmaxf(x, 0.0f) + log1pf(__expf(-fabsf(x))) - 0.69314718055994531f;
}

// ---------------- utility kernels -------------------------------------------
__global__ void zero_f32(float* __restrict__ p, int n4) {
  float4 z = {0.f, 0.f, 0.f, 0.f};
  for (int i = blockIdx.x * blockDim.x + threadIdx.x; i < n4; i += gridDim.x * blockDim.x)
    ((float4*)p)[i] = z;
}
// out[n*K + k] = (bf16) in[k*N + n]   (transpose + convert)
__global__ void cvt_transpose_bf16(const float* __restrict__ in, __bf16* __restrict__ out,
                                   int K, int N) {
  int t = blockIdx.x * blockDim.x + threadIdx.x;
  if (t >= K * N) return;
  int n = t / K, k = t - n * K;
  out[n * K + k] = (__bf16)in[k * N + n];
}

// ---------------- H = electrons @ h_w  (8192x256 @ 256x256) -----------------
__global__ void __launch_bounds__(256)
hgemm_kernel(const float* __restrict__ X, const __bf16* __restrict__ Wt, float* __restrict__ Y) {
  int lane = threadIdx.x & 31, wid = threadIdx.x >> 5;
  int half = lane >> 4, l16 = lane & 15;
  int rowBase = blockIdx.x * 128 + wid * 16;
  const float* arow = X + (size_t)(rowBase + l16) * EMB;
  v16bf af[8];
#pragma unroll
  for (int ks = 0; ks < 8; ++ks) af[ks] = a_frag_f32(arow, 32 * ks, half);
#pragma unroll 1
  for (int nt = 0; nt < 16; ++nt) {
    int col = nt * 16 + l16;
    const __bf16* brow = Wt + (size_t)col * EMB;
    v8f c = {};
#pragma unroll
    for (int ks = 0; ks < 8; ++ks) c = wmma_bf16(af[ks], b_frag_bf16t(brow, 32 * ks, half), c);
#pragma unroll
    for (int j = 0; j < 8; ++j) Y[(size_t)(rowBase + j + 8 * half) * KER + col] = c[j];
  }
}

// ---------------- fused edge pipeline ---------------------------------------
// per wave: 16 edges. we = ssp(dist@w1+b1)@w2 ; z[recv] += we * srcFeat[send]
// Weights staged once per block in LDS (116 KB total of 320 KB/WGP on CDNA5),
// so all WMMA B-fragments come from ds_load_b128 instead of L2.
__global__ void __launch_bounds__(256)
edge_kernel(const float* __restrict__ dist,       // [E, 64]
            const __bf16* __restrict__ w1t,       // [128][64]  (W1^T bf16)
            const float* __restrict__ b1,         // [128]
            const __bf16* __restrict__ w2t,       // [256][128] (W2^T bf16)
            const int* __restrict__ senders,
            const int* __restrict__ receivers,
            const float* __restrict__ srcFeat,    // H or nuclei, [*,256]
            float* __restrict__ z) {              // [8192,256] accumulators
  __shared__ __bf16 w1s[MID * DF];                // 16 KB
  __shared__ __bf16 w2s[KER * MID];               // 64 KB
  __shared__ float  b1s[MID];                     // 0.5 KB
  __shared__ __bf16 act[8][16][MID];              // 32 KB: per-wave activations
  __shared__ int sIdx[8][16];
  __shared__ int rIdx[8][16];

  int lane = threadIdx.x & 31, wid = threadIdx.x >> 5;
  int half = lane >> 4, l16 = lane & 15;
  int eBase = blockIdx.x * 128 + wid * 16;

  if (half == 0) sIdx[wid][l16] = senders[eBase + l16];
  else           rIdx[wid][l16] = receivers[eBase + l16];

  // ---- stage weights into LDS (cooperative 16B copies) ----
  {
    const uint4* s1 = (const uint4*)w1t;  uint4* d1 = (uint4*)w1s;
#pragma unroll
    for (int i = 0; i < MID * DF / 8 / 256; ++i)
      d1[threadIdx.x + i * 256] = s1[threadIdx.x + i * 256];
    const uint4* s2 = (const uint4*)w2t;  uint4* d2 = (uint4*)w2s;
#pragma unroll
    for (int i = 0; i < KER * MID / 8 / 256; ++i)
      d2[threadIdx.x + i * 256] = s2[threadIdx.x + i * 256];
    if (threadIdx.x < MID) b1s[threadIdx.x] = b1[threadIdx.x];
  }

  // A fragments from dist (global, fp32 -> bf16) overlap with staging
  const float* drow = dist + (size_t)(eBase + l16) * DF;
  v16bf a0 = a_frag_f32(drow, 0, half);
  v16bf a1 = a_frag_f32(drow, 32, half);
  __syncthreads();

  // ---- GEMM1: [16,64] @ [64,128] + bias -> SSP -> act (bf16 in LDS) ----
#pragma unroll 1
  for (int nt = 0; nt < 8; ++nt) {
    int col = nt * 16 + l16;
    const __bf16* brow = w1s + col * DF;
    v8f c = {};
    c = wmma_bf16(a0, b_frag_bf16t(brow, 0, half), c);
    c = wmma_bf16(a1, b_frag_bf16t(brow, 32, half), c);
    float bias = b1s[col];
#pragma unroll
    for (int j = 0; j < 8; ++j) act[wid][j + 8 * half][col] = (__bf16)ssp(c[j] + bias);
  }
  // LDS is in-order within a wave; act region is wave-private -> no barrier.

  // ---- GEMM2: [16,128] @ [128,256] with fused gather * scatter-add epilogue
  const __bf16* arow = &act[wid][l16][0];
  v16bf af[4];
#pragma unroll
  for (int ks = 0; ks < 4; ++ks) af[ks] = a_frag_bf16(arow, 32 * ks, half);
#pragma unroll 1
  for (int nt = 0; nt < 16; ++nt) {
    int col = nt * 16 + l16;
    const __bf16* brow = w2s + col * MID;
    v8f c = {};
#pragma unroll
    for (int ks = 0; ks < 4; ++ks) c = wmma_bf16(af[ks], b_frag_bf16t(brow, 32 * ks, half), c);
#pragma unroll
    for (int j = 0; j < 8; ++j) {
      int s  = sIdx[wid][j + 8 * half];
      int rc = rIdx[wid][j + 8 * half];
      float hv = srcFeat[(size_t)s * KER + col];
      atomicAdd(&z[(size_t)rc * KER + col], c[j] * hv);
    }
  }
}

// ---------------- out = electrons + z_n@g_n + z_same@g_same + z_anti@g_anti -
__global__ void __launch_bounds__(256)
out_kernel(const float* __restrict__ electrons,
           const float* __restrict__ z0, const float* __restrict__ z1,
           const float* __restrict__ z2,
           const __bf16* __restrict__ g0, const __bf16* __restrict__ g1,
           const __bf16* __restrict__ g2,
           float* __restrict__ out) {
  int lane = threadIdx.x & 31, wid = threadIdx.x >> 5;
  int half = lane >> 4, l16 = lane & 15;
  int rowBase = blockIdx.x * 128 + wid * 16;
  const float*  zs[3] = { z0, z1, z2 };
  const __bf16* gs[3] = { g0, g1, g2 };
  v8f acc[16];
#pragma unroll
  for (int nt = 0; nt < 16; ++nt) acc[nt] = (v8f){};
#pragma unroll
  for (int s = 0; s < 3; ++s) {
    const float* arow = zs[s] + (size_t)(rowBase + l16) * KER;
    v16bf af[8];
#pragma unroll
    for (int ks = 0; ks < 8; ++ks) af[ks] = a_frag_f32(arow, 32 * ks, half);
#pragma unroll
    for (int nt = 0; nt < 16; ++nt) {
      const __bf16* brow = gs[s] + (size_t)(nt * 16 + l16) * KER;
#pragma unroll
      for (int ks = 0; ks < 8; ++ks)
        acc[nt] = wmma_bf16(af[ks], b_frag_bf16t(brow, 32 * ks, half), acc[nt]);
    }
  }
#pragma unroll
  for (int nt = 0; nt < 16; ++nt) {
    int col = nt * 16 + l16;
#pragma unroll
    for (int j = 0; j < 8; ++j) {
      size_t idx = (size_t)(rowBase + j + 8 * half) * EMB + col;
      out[idx] = electrons[idx] + acc[nt][j];
    }
  }
}

// ---------------- host side --------------------------------------------------
extern "C" void kernel_launch(void* const* d_in, const int* in_sizes, int n_in,
                              void* d_out, int out_size, void* d_ws, size_t ws_size,
                              hipStream_t stream) {
  (void)in_sizes; (void)n_in; (void)out_size; (void)ws_size;
  const float* electrons = (const float*)d_in[0];
  const float* nuclei    = (const float*)d_in[1];
  const float* dist[3]   = {(const float*)d_in[2],  (const float*)d_in[7],  (const float*)d_in[12]};
  const float* w1[3]     = {(const float*)d_in[3],  (const float*)d_in[8],  (const float*)d_in[13]};
  const float* b1[3]     = {(const float*)d_in[4],  (const float*)d_in[9],  (const float*)d_in[14]};
  const float* w2[3]     = {(const float*)d_in[5],  (const float*)d_in[10], (const float*)d_in[15]};
  const float* g[3]      = {(const float*)d_in[6],  (const float*)d_in[11], (const float*)d_in[16]};
  const float* h_w       = (const float*)d_in[17];
  const int* senders[3]  = {(const int*)d_in[18], (const int*)d_in[19], (const int*)d_in[20]};
  const int* receivers[3]= {(const int*)d_in[21], (const int*)d_in[22], (const int*)d_in[23]};
  float* out = (float*)d_out;

  // workspace layout
  char* ws = (char*)d_ws;
  float* H    = (float*)ws;                                  // 8 MB
  float* zbuf = (float*)(ws + (size_t)8 * 1024 * 1024);      // 24 MB
  float* z[3] = { zbuf, zbuf + (size_t)N_ELEC * KER, zbuf + (size_t)2 * N_ELEC * KER };
  __bf16* wb = (__bf16*)(ws + (size_t)32 * 1024 * 1024);
  __bf16* h_wt  = wb;                                        // 65536
  __bf16* gt[3] = { wb + 65536, wb + 2 * 65536, wb + 3 * 65536 };
  __bf16* w1t[3]= { wb + 4 * 65536, wb + 4 * 65536 + 8192, wb + 4 * 65536 + 2 * 8192 };
  __bf16* w2t[3]= { wb + 4 * 65536 + 3 * 8192,
                    wb + 4 * 65536 + 3 * 8192 + 32768,
                    wb + 4 * 65536 + 3 * 8192 + 2 * 32768 };

  // 1) zero z accumulators (graph replays don't re-zero)
  zero_f32<<<512, 256, 0, stream>>>(zbuf, 3 * N_ELEC * KER / 4);

  // 2) convert+transpose weights to bf16
  cvt_transpose_bf16<<<(65536 + 255) / 256, 256, 0, stream>>>(h_w, h_wt, 256, 256);
  for (int i = 0; i < 3; ++i) {
    cvt_transpose_bf16<<<(65536 + 255) / 256, 256, 0, stream>>>(g[i],  gt[i],  256, 256);
    cvt_transpose_bf16<<<(8192  + 255) / 256, 256, 0, stream>>>(w1[i], w1t[i], 64,  128);
    cvt_transpose_bf16<<<(32768 + 255) / 256, 256, 0, stream>>>(w2[i], w2t[i], 128, 256);
  }

  // 3) H = electrons @ h_w (compute once, gather per edge afterwards)
  hgemm_kernel<<<N_ELEC / 128, 256, 0, stream>>>(electrons, h_wt, H);

  // 4) fused edge pipelines (same, anti use H; n uses nuclei directly)
  edge_kernel<<<E_EDGES / 128, 256, 0, stream>>>(dist[0], w1t[0], b1[0], w2t[0],
                                                 senders[0], receivers[0], H, z[0]);
  edge_kernel<<<E_EDGES / 128, 256, 0, stream>>>(dist[1], w1t[1], b1[1], w2t[1],
                                                 senders[1], receivers[1], H, z[1]);
  edge_kernel<<<E_EDGES / 128, 256, 0, stream>>>(dist[2], w1t[2], b1[2], w2t[2],
                                                 senders[2], receivers[2], nuclei, z[2]);

  // 5) out = electrons + z_n@g_n + z_same@g_same + z_anti@g_anti
  out_kernel<<<N_ELEC / 128, 256, 0, stream>>>(electrons, z[2], z[0], z[1],
                                               gt[2], gt[0], gt[1], out);
}
